// VanillaRNN_35235911696455
// MI455X (gfx1250) — matrix operand order repaired and checked by reading, not compiled, not run on hardware
//
#include <hip/hip_runtime.h>
#include <hip/hip_bf16.h>
#include <stdint.h>

// VanillaRNN for MI455X (gfx1250): bf16 WMMA GEMMs + persistent-WGP recurrent scan.
typedef __bf16 bf16_t;
typedef __attribute__((ext_vector_type(16))) __bf16 v16bf;
typedef __attribute__((ext_vector_type(8)))  __bf16 v8bf;
typedef __attribute__((ext_vector_type(8)))  float  v8f;

#define T_ 512
#define B_ 16
#define E_ 64
#define H_ 512
#define V_ 32000

static __device__ __forceinline__ bf16_t f32_to_bf16(float f) {
  unsigned u = __float_as_uint(f);
  u += 0x7FFFu + ((u >> 16) & 1u);            // round-to-nearest-even
  unsigned short s = (unsigned short)(u >> 16);
  bf16_t b;
  __builtin_memcpy(&b, &s, sizeof(s));
  return b;
}

union V16 { v16bf v; v8bf h[2]; };

// A fragment (16x32 bf16, MxK). Layout: lanes 0-15 hold M=lane,
// K=k0..k0+15; lanes 16-31 hold K=k0+16..k0+31 (matches B fragment).
static __device__ __forceinline__ v16bf load_a16x32(const bf16_t* A, int lda, int k0, int lane) {
  int m  = lane & 15;
  int kb = k0 + ((lane & 16) ? 16 : 0);
  const bf16_t* p = A + (size_t)m * lda + kb;
  V16 u;
  u.h[0] = *(const v8bf*)(p);
  u.h[1] = *(const v8bf*)(p + 8);
  return u.v;
}

// B fragment (32x16 bf16, KxN) taken from W[N,K] row-major (so contiguous in K per lane).
// Layout used here (contiguous k, same as A): lane gives N=lane&15; lanes 0-15 hold K=k0..k0+15, lanes 16-31 K=k0+16..k0+31.
static __device__ __forceinline__ v16bf load_b32x16(const bf16_t* W, int ldw, int n0, int k0, int lane) {
  int n  = n0 + (lane & 15);
  int kb = k0 + ((lane & 16) ? 16 : 0);
  const bf16_t* p = W + (size_t)n * ldw + kb;
  V16 u;
  u.h[0] = *(const v8bf*)(p);
  u.h[1] = *(const v8bf*)(p + 8);
  return u.v;
}

static __device__ __forceinline__ v8f wmma_bf16(v16bf a, v16bf b, v8f c) {
  return __builtin_amdgcn_wmma_f32_16x16x32_bf16(false, a, false, b, (short)0, c, false, false);
}

// ---------------------------------------------------------------------------
// fp32 -> bf16 weight conversion
__global__ void k_f32_to_bf16(const float* __restrict__ src, bf16_t* __restrict__ dst, int n) {
  int i = blockIdx.x * blockDim.x + threadIdx.x;
  if (i < n) dst[i] = f32_to_bf16(src[i]);
}

// ---------------------------------------------------------------------------
// Fused embedding lookup + input projection: xh[t][b][h] = emb[x[b][t]] . W_xh[h] + b_xh[h]
__global__ void k_embed_proj(const int* __restrict__ x, const float* __restrict__ emb,
                             const float* __restrict__ W_xh, const float* __restrict__ b_xh,
                             float* __restrict__ xh) {
  int gid = blockIdx.x * blockDim.x + threadIdx.x;   // (t*B+b)*H + h
  int h  = gid & (H_ - 1);
  int tb = gid >> 9;
  int t = tb >> 4, b = tb & 15;
  int tok = x[b * T_ + t];
  const float* e = emb  + (size_t)tok * E_;
  const float* w = W_xh + (size_t)h * E_;
  float acc = b_xh[h];
  #pragma unroll
  for (int k = 0; k < E_; ++k) acc += e[k] * w[k];
  xh[gid] = acc;
}

// ---------------------------------------------------------------------------
// Persistent recurrent scan: 1 workgroup, 16 waves on one WGP.
// Per step: h_new = LN(tanh(xh_t + h_prev @ W_hh^T + b_hh)); h kept in LDS (bf16).
// Half of W_hh (k<256) is held register-resident across all 512 steps (16 frags =
// 128 VGPRs/wave); the other half streams from L2 each step. The streamed loads
// index through an opaque zero VGPR offset: this defeats loop-invariant hoisting
// (no scratch spills) while preserving pointer provenance so they lower to
// global_load (LOADcnt only), not flat_load (which would couple to DScnt and
// serialize against the LDS A-fragment reads).
__global__ void __launch_bounds__(512, 1)
k_rnn_scan(const float* __restrict__ xh, const bf16_t* __restrict__ Whh,
           const float* __restrict__ b_hh, const float* __restrict__ gamma,
           const float* __restrict__ beta, bf16_t* __restrict__ hs) {
  __shared__ __align__(16) bf16_t h_bf[B_ * H_];   // 16 KB: h_prev as WMMA A operand
  __shared__ __align__(16) float  a_f[B_ * H_];    // 32 KB: pre-LN activations

  const int tid  = threadIdx.x;
  const int lane = tid & 31;
  const int w    = tid >> 5;     // wave id 0..15
  const int n0   = w * 32;       // this wave owns output columns n0..n0+31

  // Register-resident B fragments for k in [0,256): 2 tiles x 8 k-steps.
  v16bf Bc0[8], Bc1[8];
  #pragma unroll
  for (int kk = 0; kk < 8; ++kk) {
    Bc0[kk] = load_b32x16(Whh, H_, n0,      kk * 32, lane);
    Bc1[kk] = load_b32x16(Whh, H_, n0 + 16, kk * 32, lane);
  }

  for (int i = tid; i < B_ * H_; i += 512) h_bf[i] = f32_to_bf16(0.f);
  __syncthreads();

  for (int t = 0; t < T_; ++t) {
    v8f acc0 = {}; v8f acc1 = {};
    // k in [0,256): weights from registers, A from LDS.
    #pragma unroll
    for (int kk = 0; kk < 8; ++kk) {
      v16bf a = load_a16x32(h_bf, H_, kk * 32, lane);
      acc0 = wmma_bf16(a, Bc0[kk], acc0);
      acc1 = wmma_bf16(a, Bc1[kk], acc1);
    }
    // k in [256,512): weights streamed from L2 every step.
    {
      int zofs = 0;
      asm volatile("" : "+v"(zofs));         // opaque 0: defeats hoisting, keeps
      const bf16_t* wp = Whh + zofs;         // global addrspace provenance
      #pragma unroll
      for (int kk = 8; kk < 16; ++kk) {
        v16bf a  = load_a16x32(h_bf, H_, kk * 32, lane);
        v16bf b0 = load_b32x16(wp, H_, n0,      kk * 32, lane);
        v16bf b1 = load_b32x16(wp, H_, n0 + 16, kk * 32, lane);
        acc0 = wmma_bf16(a, b0, acc0);
        acc1 = wmma_bf16(a, b1, acc1);
      }
    }
    // + xh_t + b_hh, tanh -> a_f
    const float* xh_t = xh + (size_t)t * (B_ * H_);
    const int mo = (lane & 16) ? 8 : 0;
    const int c0 = n0 + (lane & 15);
    const int c1 = c0 + 16;
    #pragma unroll
    for (int r = 0; r < 8; ++r) {
      int b = mo + r;
      a_f[b * H_ + c0] = tanhf(acc0[r] + xh_t[b * H_ + c0] + b_hh[c0]);
      a_f[b * H_ + c1] = tanhf(acc1[r] + xh_t[b * H_ + c1] + b_hh[c1]);
    }
    __syncthreads();

    // LayerNorm: wave w owns batch row b=w (512 elements), wave32 shuffle reduction.
    {
      const float* row = a_f + w * H_;
      float s = 0.f, s2 = 0.f;
      #pragma unroll
      for (int i = 0; i < H_ / 32; ++i) {
        float v = row[lane + 32 * i];
        s += v; s2 += v * v;
      }
      #pragma unroll
      for (int off = 16; off > 0; off >>= 1) {
        s  += __shfl_xor(s,  off, 32);
        s2 += __shfl_xor(s2, off, 32);
      }
      float mu  = s * (1.0f / H_);
      float var = s2 * (1.0f / H_) - mu * mu;
      float rs  = rsqrtf(var + 1e-5f);
      bf16_t* hrow = h_bf + w * H_;
      bf16_t* orow = hs + (size_t)t * (B_ * H_) + w * H_;
      #pragma unroll
      for (int i = 0; i < H_ / 32; ++i) {
        int j = lane + 32 * i;
        bf16_t bb = f32_to_bf16((row[j] - mu) * rs * gamma[j] + beta[j]);
        hrow[j] = bb;       // carry for next step (WMMA A operand)
        orow[j] = bb;       // hs[t][b][:] for output GEMM
      }
    }
    __syncthreads();
  }
}

// ---------------------------------------------------------------------------
// Output projection: out[b][t][v] = hs[t][b][:] . W_ho[v][:] + b_ho[v]
// M=(t*16+b)=8192, N=32000, K=512.  8 waves/WG; wave tile 16x64; WG tile 32x256.
// Logits are streamed out with non-temporal stores (1 GB, write-once) so the
// L2 stays dedicated to W_ho (32 MB bf16, fully L2-resident).
__global__ void __launch_bounds__(256, 2)
k_out_gemm(const bf16_t* __restrict__ hs, const bf16_t* __restrict__ Who,
           const float* __restrict__ b_ho, float* __restrict__ out) {
  const int lane = threadIdx.x & 31;
  const int w    = threadIdx.x >> 5;   // 0..7
  const int wm   = w >> 2;             // 0..1
  const int wn   = w & 3;              // 0..3
  const int m0 = blockIdx.y * 32 + wm * 16;
  const int n0 = blockIdx.x * 256 + wn * 64;

  const bf16_t* A = hs + (size_t)m0 * H_;
  v8f acc[4] = {{}, {}, {}, {}};
  for (int k0 = 0; k0 < H_; k0 += 32) {
    v16bf a = load_a16x32(A, H_, k0, lane);      // reused across 4 N tiles
    #pragma unroll
    for (int j = 0; j < 4; ++j) {
      v16bf b = load_b32x16(Who, H_, n0 + j * 16, k0, lane);
      acc[j] = wmma_bf16(a, b, acc[j]);
    }
    if (k0 + 32 < H_) {
      __builtin_prefetch(Who + (size_t)(n0 + (lane & 15)) * H_ + k0 + 32, 0, 1);
    }
  }
  const int mo = (lane & 16) ? 8 : 0;
  #pragma unroll
  for (int j = 0; j < 4; ++j) {
    int n = n0 + j * 16 + (lane & 15);
    float bias = b_ho[n];
    #pragma unroll
    for (int r = 0; r < 8; ++r) {
      int m = m0 + mo + r;
      int t = m >> 4, b = m & 15;
      __builtin_nontemporal_store(acc[j][r] + bias,
                                  out + ((size_t)b * T_ + t) * V_ + n);
    }
  }
}

// ---------------------------------------------------------------------------
extern "C" void kernel_launch(void* const* d_in, const int* in_sizes, int n_in,
                              void* d_out, int out_size, void* d_ws, size_t ws_size,
                              hipStream_t stream) {
  const int*   x     = (const int*)d_in[0];
  const float* emb   = (const float*)d_in[1];
  const float* W_xh  = (const float*)d_in[2];
  const float* b_xh  = (const float*)d_in[3];
  const float* W_hh  = (const float*)d_in[4];
  const float* b_hh  = (const float*)d_in[5];
  const float* W_ho  = (const float*)d_in[6];
  const float* b_ho  = (const float*)d_in[7];
  const float* gamma = (const float*)d_in[8];
  const float* beta  = (const float*)d_in[9];
  float* out = (float*)d_out;

  // Workspace layout (all 256B aligned):
  char* ws = (char*)d_ws;
  float*  xh    = (float*)(ws);                                      // 16 MB  [T][B][H] fp32
  bf16_t* Whh_b = (bf16_t*)(ws + 16777216);                          // 512 KB [H][H] bf16
  bf16_t* Who_b = (bf16_t*)(ws + 16777216 + 524288);                 // 32 MB  [V][H] bf16
  bf16_t* hs_b  = (bf16_t*)(ws + 16777216 + 524288 + 32768000);      // 8 MB   [T][B][H] bf16

  k_f32_to_bf16<<<(H_ * H_ + 255) / 256, 256, 0, stream>>>(W_hh, Whh_b, H_ * H_);
  k_f32_to_bf16<<<(V_ * H_ + 255) / 256, 256, 0, stream>>>(W_ho, Who_b, V_ * H_);
  k_embed_proj <<<(T_ * B_ * H_) / 256, 256, 0, stream>>>(x, emb, W_xh, b_xh, xh);
  k_rnn_scan   <<<1, 512, 0, stream>>>(xh, Whh_b, b_hh, gamma, beta, hs_b);
  k_out_gemm   <<<dim3(V_ / 256, (T_ * B_) / 32), 256, 0, stream>>>(hs_b, Who_b, b_ho, out);
}
